// ProteinLigandAttention_45535243272653
// MI455X (gfx1250) — matrix operand order repaired and checked by reading, not compiled, not run on hardware
//
#include <hip/hip_runtime.h>

// ---------------------------------------------------------------------------
// ProteinLigandAttention on MI455X (gfx1250), wave32, WMMA bf16 path.
//
// Pipeline (all on `stream`):
//   1) pl_dist_kernel: dists[4096][1024] = cdist(protein_coords, ligand_coords)
//   2) pl_cvt_bf16 x6: pre-convert features + weights to bf16 (once)
//   3) pl_gemm256_bf16 x6: Qp, Kl, Vlt (transposed), Ql, Kp, Vpt — pure bf16
//      b128 loads, 8 x v_wmma_f32_16x16x32_bf16 per 16x16 tile
//   4) pl_attn_wmma x2: flash-attention, 1 wave/head, 16-query tiles,
//      32-key chunks, online softmax, dist-bias + mask; bf16 output
//   5) pl_gemm256_bf16 x2 (mode 2): output projection -> d_out (f32)
// ---------------------------------------------------------------------------

#define NP 4096
#define NL 1024
#define FD 256
#define NHEAD 8
#define HDIM 32
#define ATT_SCALE 0.17677669529663687f   // 32^-0.5
#define NEG_BIG (-1.0e30f)

typedef __bf16 bf16_t;
typedef bf16_t bf16x16 __attribute__((ext_vector_type(16)));
typedef bf16_t bf16x2  __attribute__((ext_vector_type(2)));
typedef float  f32x8   __attribute__((ext_vector_type(8)));

struct alignas(16) U4 { unsigned int x, y, z, w; };
struct alignas(16) F4 { float x, y, z, w; };

union FragB16 {
  bf16x16        v;     // operand for v_wmma_f32_16x16x32_bf16
  unsigned short s[16];
  U4             q[2];
};

__device__ __forceinline__ unsigned short f2bf(float f) {
  unsigned int u = __float_as_uint(f);
  u += 0x7fffu + ((u >> 16) & 1u);      // round-to-nearest-even
  return (unsigned short)(u >> 16);
}

#if defined(__has_builtin)
#if __has_builtin(__builtin_amdgcn_cvt_pk_bf16_f32)
#define HAVE_CVT_PK_BF16 1
#endif
#endif

__device__ __forceinline__ unsigned int pack2bf(float a, float b) {
#ifdef HAVE_CVT_PK_BF16
  union { bf16x2 v; unsigned int u; } c;
  c.v = __builtin_amdgcn_cvt_pk_bf16_f32(a, b);
  return c.u;
#else
  return (unsigned int)f2bf(a) | ((unsigned int)f2bf(b) << 16);
#endif
}

// ---------------------------------------------------------------------------
// 1) Pairwise distances (cdist), computed once, reused by all heads x 2 dirs.
// ---------------------------------------------------------------------------
__global__ void pl_dist_kernel(const float* __restrict__ pc,
                               const float* __restrict__ lc,
                               float* __restrict__ dists) {
  int idx = blockIdx.x * blockDim.x + threadIdx.x;
  if (idx >= NP * NL) return;
  int i = idx >> 10;        // / NL
  int j = idx & (NL - 1);   // % NL
  float dx = pc[i * 3 + 0] - lc[j * 3 + 0];
  float dy = pc[i * 3 + 1] - lc[j * 3 + 1];
  float dz = pc[i * 3 + 2] - lc[j * 3 + 2];
  dists[idx] = sqrtf(dx * dx + dy * dy + dz * dz);
}

// ---------------------------------------------------------------------------
// 2) Bulk f32 -> bf16 conversion (8 elements/thread, b128 in / b128 out).
// ---------------------------------------------------------------------------
__global__ void pl_cvt_bf16(const float* __restrict__ src,
                            unsigned short* __restrict__ dst, int n) {
  int base = (blockIdx.x * blockDim.x + threadIdx.x) * 8;
  if (base >= n) return;
  const F4* s = (const F4*)(src + base);
  F4 a = s[0], b = s[1];
  U4 o;
  o.x = pack2bf(a.x, a.y);
  o.y = pack2bf(a.z, a.w);
  o.z = pack2bf(b.x, b.y);
  o.w = pack2bf(b.z, b.w);
  *(U4*)(dst + base) = o;
}

// ---------------------------------------------------------------------------
// 3/5) GEMM: out[M,256] = A[M,256](bf16) @ W[256,256]^T(bf16) + bias(f32)
// One wave per 16x16 output tile; K-loop of 8 x v_wmma_f32_16x16x32_bf16.
// mode 0: bf16 row-major   mode 1: bf16 transposed [256][M]   mode 2: f32 rm
//
// Fragment layouts (CDNA5 ISA 7.12.2, 16-bit, wave32):
//   A 16x32: lane l -> row = l%16; halfs j<8: K = 8*(l/16)+j,
//                                  j>=8:    K = 16 + 8*(l/16) + (j-8)
//   B 32x16: lane l -> col = l%16; same K mapping as A
//   C 16x16: vgpr r, lane l -> row = r + 8*(l/16), col = l%16
// All A/B fetches are contiguous 128-bit loads against row-major A / W.
// ---------------------------------------------------------------------------
__global__ void pl_gemm256_bf16(const unsigned short* __restrict__ A,
                                const unsigned short* __restrict__ W,
                                const float* __restrict__ bias,
                                void* __restrict__ out,
                                int M, int mode) {
  int wave = blockIdx.x * (blockDim.x >> 5) + (threadIdx.x >> 5);
  if (wave >= M) return;                  // tiles == (M/16)*(256/16) == M
  int lane = threadIdx.x & 31;
  int nl = lane & 15, hi = lane >> 4;
  int tile_m = wave >> 4;
  int tile_n = wave & 15;
  int row  = tile_m * 16 + nl;            // A-frag row (M index)
  int ncol = tile_n * 16 + nl;            // B-frag col (N index)

  const unsigned short* arow = A + (size_t)row * FD;
  const unsigned short* wrow = W + (size_t)ncol * FD;

  f32x8 acc = {};
#pragma unroll
  for (int kb = 0; kb < 8; ++kb) {
    int k0 = kb * 32;
    FragB16 af, bf;
    af.q[0] = *(const U4*)(arow + k0 + 8 * hi);
    af.q[1] = *(const U4*)(arow + k0 + 16 + 8 * hi);
    const U4* bp = (const U4*)(wrow + k0 + 8 * hi);
    bf.q[0] = bp[0];
    bf.q[1] = bp[2];
    acc = __builtin_amdgcn_wmma_f32_16x16x32_bf16(
        false, af.v, false, bf.v, (short)0, acc, false, false);
  }

  float bv = bias[ncol];
#pragma unroll
  for (int r = 0; r < 8; ++r) {
    int orow = tile_m * 16 + r + 8 * hi;
    float val = acc[r] + bv;
    if (mode == 0) {
      ((unsigned short*)out)[(size_t)orow * FD + ncol] = f2bf(val);
    } else if (mode == 1) {
      ((unsigned short*)out)[(size_t)ncol * M + orow] = f2bf(val);   // V^T
    } else {
      ((float*)out)[(size_t)orow * FD + ncol] = val;
    }
  }
}

// ---------------------------------------------------------------------------
// 4) Fused attention: one wave per head, 16 query rows per block,
//    32-key chunks with online softmax. dist/mask indexed with strides so the
//    same kernel serves both directions (transposed views).
//    Q    : bf16 row-major [Nq][256]
//    Kf   : bf16 row-major [Nk][256]
//    Vt   : bf16 transposed [256][Nk]
//    Oattn: bf16 [Nq][256] (pre output-projection; fed back into the GEMM)
// ---------------------------------------------------------------------------
__global__ void pl_attn_wmma(const unsigned short* __restrict__ Q,
                             const unsigned short* __restrict__ Kf,
                             const unsigned short* __restrict__ Vt,
                             const float* __restrict__ dists,
                             const unsigned char* __restrict__ mask,
                             const float* __restrict__ wd,
                             const float* __restrict__ bd,
                             unsigned short* __restrict__ Oattn,
                             int Nq, int Nk, int d_qs, int d_ks) {
  __shared__ unsigned short Plds[NHEAD * 16 * 32];   // 8 KB, per-wave regions
  int h    = threadIdx.x >> 5;
  int lane = threadIdx.x & 31;
  int nl = lane & 15, hi = lane >> 4;
  int qbase = blockIdx.x * 16;
  float wdh = wd[h], bdh = bd[h];

  // Q fragment (A layout), resident for the whole key loop.
  FragB16 qf;
  const unsigned short* qrow = Q + (size_t)(qbase + nl) * FD + h * HDIM;
  qf.q[0] = *(const U4*)(qrow + 8 * hi);
  qf.q[1] = *(const U4*)(qrow + 16 + 8 * hi);

  float m_i[8], l_i[8];
  f32x8 accA = {}, accB = {};
#pragma unroll
  for (int r = 0; r < 8; ++r) { m_i[r] = NEG_BIG; l_i[r] = 0.f; }

  f32x8 zero = {};
  for (int kbase = 0; kbase < Nk; kbase += 32) {
    if (kbase + 32 < Nk)
      __builtin_prefetch(Kf + (size_t)(kbase + 32 + nl) * FD + h * HDIM, 0, 0);

    float s0[8], s1[8];
    // ---- scores, key sub-block 0 (keys kbase..kbase+15) ----
    {
      FragB16 kfrag;
      const unsigned short* kr =
          Kf + (size_t)(kbase + nl) * FD + h * HDIM + 8 * hi;
      kfrag.q[0] = ((const U4*)kr)[0];
      kfrag.q[1] = ((const U4*)kr)[2];
      f32x8 Sv = __builtin_amdgcn_wmma_f32_16x16x32_bf16(
          false, qf.v, false, kfrag.v, (short)0, zero, false, false);
      int key = kbase + nl;
#pragma unroll
      for (int r = 0; r < 8; ++r) {
        size_t dm = (size_t)(qbase + r + 8 * hi) * d_qs + (size_t)key * d_ks;
        float v = Sv[r] * ATT_SCALE + dists[dm] * wdh + bdh;
        s0[r] = mask[dm] ? v : NEG_BIG;
      }
    }
    // ---- scores, key sub-block 1 (keys kbase+16..kbase+31) ----
    {
      FragB16 kfrag;
      const unsigned short* kr =
          Kf + (size_t)(kbase + 16 + nl) * FD + h * HDIM + 8 * hi;
      kfrag.q[0] = ((const U4*)kr)[0];
      kfrag.q[1] = ((const U4*)kr)[2];
      f32x8 Sv = __builtin_amdgcn_wmma_f32_16x16x32_bf16(
          false, qf.v, false, kfrag.v, (short)0, zero, false, false);
      int key = kbase + 16 + nl;
#pragma unroll
      for (int r = 0; r < 8; ++r) {
        size_t dm = (size_t)(qbase + r + 8 * hi) * d_qs + (size_t)key * d_ks;
        float v = Sv[r] * ATT_SCALE + dists[dm] * wdh + bdh;
        s1[r] = mask[dm] ? v : NEG_BIG;
      }
    }

    // ---- online softmax; row lives across 16 lanes of this half-wave ----
#pragma unroll
    for (int r = 0; r < 8; ++r) {
      float mx = fmaxf(s0[r], s1[r]);
      mx = fmaxf(mx, __shfl_xor(mx, 1, 32));
      mx = fmaxf(mx, __shfl_xor(mx, 2, 32));
      mx = fmaxf(mx, __shfl_xor(mx, 4, 32));
      mx = fmaxf(mx, __shfl_xor(mx, 8, 32));
      float mnew = fmaxf(m_i[r], mx);
      float sc = __expf(m_i[r] - mnew);
      m_i[r] = mnew;
      float p0 = __expf(s0[r] - mnew);
      float p1 = __expf(s1[r] - mnew);
      float ps = p0 + p1;
      ps += __shfl_xor(ps, 1, 32);
      ps += __shfl_xor(ps, 2, 32);
      ps += __shfl_xor(ps, 4, 32);
      ps += __shfl_xor(ps, 8, 32);
      l_i[r] = l_i[r] * sc + ps;
      accA[r] *= sc;
      accB[r] *= sc;
      // P tile (C-layout element) -> LDS, bf16
      Plds[h * 512 + (r + 8 * hi) * 32 + nl]      = f2bf(p0);
      Plds[h * 512 + (r + 8 * hi) * 32 + 16 + nl] = f2bf(p1);
    }

    // C-layout -> A-layout via per-wave LDS region; explicit CDNA5 DS wait.
    asm volatile("s_wait_dscnt 0" ::: "memory");

    FragB16 pfrag;
    const unsigned short* pl = &Plds[h * 512 + nl * 32];
    pfrag.q[0] = *(const U4*)(pl + 8 * hi);
    pfrag.q[1] = *(const U4*)(pl + 16 + 8 * hi);

    // ---- O += P @ V, dims split into two 16-wide WMMAs ----
    {
      FragB16 vf;
      const unsigned short* vr =
          Vt + (size_t)(h * HDIM + nl) * Nk + kbase + 8 * hi;
      vf.q[0] = ((const U4*)vr)[0];
      vf.q[1] = ((const U4*)vr)[2];
      accA = __builtin_amdgcn_wmma_f32_16x16x32_bf16(
          false, pfrag.v, false, vf.v, (short)0, accA, false, false);
    }
    {
      FragB16 vf;
      const unsigned short* vr =
          Vt + (size_t)(h * HDIM + 16 + nl) * Nk + kbase + 8 * hi;
      vf.q[0] = ((const U4*)vr)[0];
      vf.q[1] = ((const U4*)vr)[2];
      accB = __builtin_amdgcn_wmma_f32_16x16x32_bf16(
          false, pfrag.v, false, vf.v, (short)0, accB, false, false);
    }
  }

#pragma unroll
  for (int r = 0; r < 8; ++r) {
    float inv = (l_i[r] > 0.f) ? (1.0f / l_i[r]) : 0.f;
    int orow = qbase + r + 8 * hi;
    Oattn[(size_t)orow * FD + h * HDIM + nl]      = f2bf(accA[r] * inv);
    Oattn[(size_t)orow * FD + h * HDIM + 16 + nl] = f2bf(accB[r] * inv);
  }
}

// ---------------------------------------------------------------------------
// Host orchestration
// ---------------------------------------------------------------------------
extern "C" void kernel_launch(void* const* d_in, const int* in_sizes, int n_in,
                              void* d_out, int out_size, void* d_ws,
                              size_t ws_size, hipStream_t stream) {
  const float* pf = (const float*)d_in[0];
  const float* lf = (const float*)d_in[1];
  const float* pc = (const float*)d_in[2];
  const float* lc = (const float*)d_in[3];
  const unsigned char* mask = (const unsigned char*)d_in[4];
  const float* Wq = (const float*)d_in[5];
  const float* bq = (const float*)d_in[6];
  const float* Wk = (const float*)d_in[7];
  const float* bk = (const float*)d_in[8];
  const float* Wv = (const float*)d_in[9];
  const float* bv = (const float*)d_in[10];
  const float* wd = (const float*)d_in[11];
  const float* bd = (const float*)d_in[12];
  const float* Wo = (const float*)d_in[13];
  const float* bo = (const float*)d_in[14];
  float* out = (float*)d_out;

  const size_t KB = 1024, MB = 1024 * 1024;
  char* ws = (char*)d_ws;
  // workspace layout (29 MB total; fits L2)
  unsigned short* pf16 = (unsigned short*)(ws + 0);                  // 2 MB
  unsigned short* lf16 = (unsigned short*)(ws + 2 * MB);             // 0.5 MB
  unsigned short* Wq16 = (unsigned short*)(ws + 2 * MB + 512 * KB);  // 128 KB
  unsigned short* Wk16 = (unsigned short*)(ws + 2 * MB + 640 * KB);
  unsigned short* Wv16 = (unsigned short*)(ws + 2 * MB + 768 * KB);
  unsigned short* Wo16 = (unsigned short*)(ws + 2 * MB + 896 * KB);
  unsigned short* Qp   = (unsigned short*)(ws + 3 * MB);             // 2 MB
  unsigned short* Kp   = (unsigned short*)(ws + 5 * MB);             // 2 MB
  unsigned short* Vpt  = (unsigned short*)(ws + 7 * MB);             // 2 MB [256][4096]
  unsigned short* Ql   = (unsigned short*)(ws + 9 * MB);             // 0.5 MB
  unsigned short* Kl   = (unsigned short*)(ws + 9 * MB + 512 * KB);  // 0.5 MB
  unsigned short* Vlt  = (unsigned short*)(ws + 10 * MB);            // 0.5 MB [256][1024]
  float*          dists   = (float*)(ws + 10 * MB + 512 * KB);       // 16 MB
  unsigned short* Oattn_p = (unsigned short*)(ws + 26 * MB + 512 * KB); // 2 MB
  unsigned short* Oattn_l = (unsigned short*)(ws + 28 * MB + 512 * KB); // 0.5 MB

  // 1) distances
  pl_dist_kernel<<<(NP * NL) / 256, 256, 0, stream>>>(pc, lc, dists);

  // 2) one-time f32 -> bf16 conversions (8 elem/thread)
  pl_cvt_bf16<<<(NP * FD) / 2048, 256, 0, stream>>>(pf, pf16, NP * FD);
  pl_cvt_bf16<<<(NL * FD) / 2048, 256, 0, stream>>>(lf, lf16, NL * FD);
  pl_cvt_bf16<<<(FD * FD) / 2048, 256, 0, stream>>>(Wq, Wq16, FD * FD);
  pl_cvt_bf16<<<(FD * FD) / 2048, 256, 0, stream>>>(Wk, Wk16, FD * FD);
  pl_cvt_bf16<<<(FD * FD) / 2048, 256, 0, stream>>>(Wv, Wv16, FD * FD);
  pl_cvt_bf16<<<(FD * FD) / 2048, 256, 0, stream>>>(Wo, Wo16, FD * FD);

  // 3) QKV projections (grid = M waves, 4 waves / 128-thread block)
  pl_gemm256_bf16<<<NP / 4, 128, 0, stream>>>(pf16, Wq16, bq, Qp,  NP, 0);
  pl_gemm256_bf16<<<NL / 4, 128, 0, stream>>>(lf16, Wk16, bk, Kl,  NL, 0);
  pl_gemm256_bf16<<<NL / 4, 128, 0, stream>>>(lf16, Wv16, bv, Vlt, NL, 1);
  pl_gemm256_bf16<<<NL / 4, 128, 0, stream>>>(lf16, Wq16, bq, Ql,  NL, 0);
  pl_gemm256_bf16<<<NP / 4, 128, 0, stream>>>(pf16, Wk16, bk, Kp,  NP, 0);
  pl_gemm256_bf16<<<NP / 4, 128, 0, stream>>>(pf16, Wv16, bv, Vpt, NP, 1);

  // 4) attention: protein queries attend ligand; ligand queries attend protein
  pl_attn_wmma<<<NP / 16, 256, 0, stream>>>(Qp, Kl, Vlt, dists, mask, wd, bd,
                                            Oattn_p, NP, NL, /*d_qs=*/NL,
                                            /*d_ks=*/1);
  pl_attn_wmma<<<NL / 16, 256, 0, stream>>>(Ql, Kp, Vpt, dists, mask, wd, bd,
                                            Oattn_l, NL, NP, /*d_qs=*/1,
                                            /*d_ks=*/NL);

  // 5) output projections -> d_out (f32, concatenated)
  pl_gemm256_bf16<<<NP / 4, 128, 0, stream>>>(Oattn_p, Wo16, bo, out, NP, 2);
  pl_gemm256_bf16<<<NL / 4, 128, 0, stream>>>(Oattn_l, Wo16, bo,
                                              out + (size_t)NP * FD, NL, 2);
}
